// Propagation_927712936457
// MI455X (gfx1250) — matrix operand order repaired and checked by reading, not compiled, not run on hardware
//
#include <hip/hip_runtime.h>
#include <stdint.h>

#define B_ 4
#define N_ 4096
#define D_ 256

#define WAVES 8
#define THREADS (WAVES * 32)
#define ROWS_PER_BLOCK (WAVES * 16)   // 128 rows of i per workgroup
#define JT 32                         // j-tile (K of GEMM2 = one bf16 WMMA K)
#define NTILES (N_ / JT)              // 128

typedef __attribute__((ext_vector_type(16))) __bf16 v16bf;
typedef __attribute__((ext_vector_type(8)))  float  v8f;

union Frag {
  v16bf v;
  uint4 q[2];
  unsigned int ui[8];
};

struct Tile { float4 f[8]; };   // 4 rows x 8 cols per thread (256 threads)

__device__ __forceinline__ unsigned short bfbits(float f) {
  union { __bf16 h; unsigned short s; } u;
  u.h = (__bf16)f;                 // native RNE f32->bf16
  return u.s;
}
__device__ __forceinline__ unsigned int bfpack(float lo, float hi) {
  return (unsigned int)bfbits(lo) | ((unsigned int)bfbits(hi) << 16);
}
__device__ __forceinline__ float fcomp(const float4& v, int i) {
  return i == 0 ? v.x : i == 1 ? v.y : i == 2 ? v.z : v.w;
}

__global__ __launch_bounds__(THREADS, 1)
void prop_kernel(const float* __restrict__ val,
                 const float* __restrict__ state,
                 float* __restrict__ out_state,
                 float* __restrict__ out_val)
{
  // A operand kept in LDS (frees 64 persistent VGPRs/wave); j tiles double-buffered
  __shared__ unsigned short Arows[ROWS_PER_BLOCK][D_ + 8];  // 66 KB, written once
  __shared__ unsigned short Xrow[2][JT][D_ + 8];            // row-major j tiles
  __shared__ unsigned short Xt[2][D_][JT + 8];              // transposed j tiles
  __shared__ unsigned short Esc[WAVES][16][JT + 8];         // per-wave E relayout

  const int b    = blockIdx.y;
  const int tid  = threadIdx.x;
  const int wave = tid >> 5;
  const int lane = tid & 31;
  const int ln   = lane & 15;
  const int hi   = lane >> 4;

  const float* Xb = val   + (size_t)b * N_ * D_;
  const float* Sb = state + (size_t)b * N_;
  const int iblk = blockIdx.x * ROWS_PER_BLOCK;
  const int i0   = iblk + wave * 16;

  // ---- prologue: stage this block's Xi rows (128 x 256) as bf16 in LDS ----
  {
    const int rr = (tid & 31) * 4;     // 0..124
    const int cc = (tid >> 5) * 32;    // 0..224
#pragma unroll
    for (int r = 0; r < 4; ++r) {
      const float* src = Xb + (size_t)(iblk + rr + r) * D_ + cc;
#pragma unroll
      for (int q = 0; q < 32; q += 8) {
        float4 f0 = *(const float4*)(src + q);
        float4 f1 = *(const float4*)(src + q + 4);
        uint4 a;
        a.x = bfpack(f0.x, f0.y); a.y = bfpack(f0.z, f0.w);
        a.z = bfpack(f1.x, f1.y); a.w = bfpack(f1.z, f1.w);
        *(uint4*)&Arows[rr + r][cc + q] = a;
      }
    }
  }

  // cooperative j-tile staging geometry: 4 rows x 8 cols per thread
  const int rr0 = (tid & 7) * 4;
  const int c0  = (tid >> 3) * 8;

  auto fetch = [&](Tile& t, int j0) {
#pragma unroll
    for (int r = 0; r < 4; ++r) {
      const float* src = Xb + (size_t)(j0 + rr0 + r) * D_ + c0;
      t.f[r * 2 + 0] = *(const float4*)(src);
      t.f[r * 2 + 1] = *(const float4*)(src + 4);
    }
  };

  auto stage = [&](const Tile& t, int buf) {
#pragma unroll
    for (int r = 0; r < 4; ++r) {
      const float4 f0 = t.f[r*2+0], f1 = t.f[r*2+1];
      uint4 a;
      a.x = bfpack(f0.x, f0.y); a.y = bfpack(f0.z, f0.w);
      a.z = bfpack(f1.x, f1.y); a.w = bfpack(f1.z, f1.w);
      *(uint4*)&Xrow[buf][rr0 + r][c0] = a;
    }
#pragma unroll
    for (int cc = 0; cc < 8; ++cc) {
      float e0 = fcomp(t.f[0*2 + (cc >> 2)], cc & 3);
      float e1 = fcomp(t.f[1*2 + (cc >> 2)], cc & 3);
      float e2 = fcomp(t.f[2*2 + (cc >> 2)], cc & 3);
      float e3 = fcomp(t.f[3*2 + (cc >> 2)], cc & 3);
      uint2 p;
      p.x = bfpack(e0, e1);
      p.y = bfpack(e2, e3);
      *(uint2*)&Xt[buf][c0 + cc][rr0] = p;
    }
  };

  v8f zero8 = {};
  v8f acc[16];
#pragma unroll
  for (int i = 0; i < 16; ++i) acc[i] = zero8;
  float stacc[8];
#pragma unroll
  for (int r = 0; r < 8; ++r) stacc[r] = 0.0f;

  // prologue: stage tile 0 into buffer 0 (Arows barrier folded into same sync)
  Tile nxt;
  fetch(nxt, 0);
  stage(nxt, 0);
  __syncthreads();

  for (int jt = 0; jt < NTILES; ++jt) {
    const int j0  = jt * JT;
    const int cur = jt & 1;
    const bool haveNext = (jt + 1) < NTILES;

    // prefetch next tile into registers; latency hidden behind the WMMAs below
    if (haveNext) fetch(nxt, j0 + JT);

    // ---- GEMM1: S(16x32) = Xi . Xj^T  (A streamed from LDS, reused x2) ----
    v8f S[2] = {zero8, zero8};
#pragma unroll
    for (int kc = 0; kc < 8; ++kc) {
      Frag af;  // A layout: row=ln, K = kc*32 + hi*8 + {0..7}, +16
      {
        const uint4* a0 = (const uint4*)&Arows[wave * 16 + ln][kc * 32 + hi * 8];
        const uint4* a1 = (const uint4*)&Arows[wave * 16 + ln][kc * 32 + hi * 8 + 16];
        af.q[0] = a0[0];
        af.q[1] = a1[0];
      }
#pragma unroll
      for (int nsub = 0; nsub < 2; ++nsub) {
        Frag bf;  // B layout: col=ln -> row j0+nsub*16+ln, K = kc*32 + hi*8 + {0..7}, +16
        const uint4* bp = (const uint4*)&Xrow[cur][nsub * 16 + ln][kc * 32 + hi * 8];
        bf.q[0] = bp[0];
        bf.q[1] = bp[2];
        S[nsub] = __builtin_amdgcn_wmma_f32_16x16x32_bf16(false, af.v, false, bf.v,
                                                          (short)0, S[nsub], false, false);
      }
    }

    // ---- softsign + delta_state partial + stage E (C-layout -> LDS) ----
#pragma unroll
    for (int nsub = 0; nsub < 2; ++nsub) {
      const float sj = Sb[j0 + nsub * 16 + ln];
#pragma unroll
      for (int r = 0; r < 8; ++r) {
        float t = S[nsub][r];
        // softsign(t/16) with scale folded; v_rcp_f32 runs in the TRANS pipe
        float e = t * __builtin_amdgcn_rcpf(16.0f + __builtin_fabsf(t));
        stacc[r] += e * sj;
        Esc[wave][r + 8 * hi][nsub * 16 + ln] = bfbits(e);
      }
    }

    // same-wave LDS is in-order; wait + compiler barrier before relayout read
    asm volatile("s_wait_dscnt 0" ::: "memory");

    Frag Ea;  // E re-read in A layout: row=ln, K = hi*8+{0..7}, 16+hi*8+{0..7}
    {
      const uint4* p0 = (const uint4*)&Esc[wave][ln][hi * 8];
      const uint4* p1 = (const uint4*)&Esc[wave][ln][hi * 8 + 16];
      Ea.q[0] = p0[0];
      Ea.q[1] = p1[0];
    }

    // ---- GEMM2: acc(16x256) += E(16x32) . Xj(32x256) ----
#pragma unroll
    for (int dsub = 0; dsub < 16; ++dsub) {
      Frag bf;
      const uint4* bp = (const uint4*)&Xt[cur][dsub * 16 + ln][hi * 8];
      bf.q[0] = bp[0];
      bf.q[1] = bp[2];
      acc[dsub] = __builtin_amdgcn_wmma_f32_16x16x32_bf16(false, Ea.v, false, bf.v,
                                                          (short)0, acc[dsub], false, false);
    }

    // stage next tile into the other buffer; overlaps with this tile's reads
    if (haveNext) stage(nxt, cur ^ 1);
    __syncthreads();   // single barrier: publish next buffer / retire this one
  }

  // ---- epilogue: delta_val ----
  float* OV = out_val + (size_t)b * N_ * D_;
#pragma unroll
  for (int dsub = 0; dsub < 16; ++dsub) {
#pragma unroll
    for (int r = 0; r < 8; ++r) {
      OV[(size_t)(i0 + r + 8 * hi) * D_ + dsub * 16 + ln] = acc[dsub][r];
    }
  }

  // ---- epilogue: delta_state (reduce across the 16 lanes of each row group) ----
#pragma unroll
  for (int r = 0; r < 8; ++r) {
    float v = stacc[r];
    v += __shfl_xor(v, 1, 32);
    v += __shfl_xor(v, 2, 32);
    v += __shfl_xor(v, 4, 32);
    v += __shfl_xor(v, 8, 32);
    stacc[r] = v;
  }
  if (ln == 0) {  // lanes 0 and 16 hold rows 0-7 and 8-15
    float* OS = out_state + (size_t)b * N_;
#pragma unroll
    for (int r = 0; r < 8; ++r) OS[i0 + 8 * hi + r] = stacc[r];
  }
}

extern "C" void kernel_launch(void* const* d_in, const int* in_sizes, int n_in,
                              void* d_out, int out_size, void* d_ws, size_t ws_size,
                              hipStream_t stream) {
  (void)in_sizes; (void)n_in; (void)out_size; (void)d_ws; (void)ws_size;
  const float* val   = (const float*)d_in[0];   // [B, N, D]
  const float* state = (const float*)d_in[1];   // [B, N]
  float* out_state = (float*)d_out;                      // [B, N]
  float* out_val   = (float*)d_out + (size_t)B_ * N_;    // [B, N, D]
  dim3 grid(N_ / ROWS_PER_BLOCK, B_);
  dim3 block(THREADS);
  prop_kernel<<<grid, block, 0, stream>>>(val, state, out_state, out_val);
}
